// BiRNNModel_84164179132792
// MI455X (gfx1250) — matrix operand order repaired and checked by reading, not compiled, not run on hardware
//
#include <hip/hip_runtime.h>
#include <hip/hip_bf16.h>
#include <math.h>

// ---------------------------------------------------------------------------
// BiRNN (tanh) on MI455X / gfx1250.
// - bf16 WMMA (v_wmma_f32_16x16x32_bf16) with fp32 accumulation.
// - Weight (B) tiles staged to LDS via CDNA5 async loads
//   (global_load_async_to_lds_b128, ASYNCcnt), TRIPLE-buffered with one
//   workgroup barrier per K-chunk.
// - Reference's "store hf into backward slot" bug makes the backward
//   direction non-recurrent => only the two forward chains are sequential.
// ---------------------------------------------------------------------------

typedef __attribute__((ext_vector_type(16))) __bf16        v16bf;
typedef __attribute__((ext_vector_type(8)))  float          v8f;
typedef __attribute__((ext_vector_type(4)))  unsigned int   u32x4;
typedef __attribute__((ext_vector_type(4)))  float          f32x4;
typedef __attribute__((ext_vector_type(4)))  unsigned short u16x4;

#define RNN_K 1024             // D == H == 1024
#define RNN_B 64               // batch
#define RNN_S 256              // sequence length
#define RNN_M (RNN_S * RNN_B)  // 16384 rows in the "big" GEMMs
#define KCHUNKS (RNN_K / 32)   // 32 K-chunks of 32
#define BTILE_H (64 * 32)      // halves per staged B tile (4KB)

__device__ __forceinline__ unsigned short f2bf(float f) {
  union { float f; unsigned u; } v; v.f = f;
  unsigned u = v.u;
  unsigned r = u + 0x7FFFu + ((u >> 16) & 1u);   // round-to-nearest-even
  return (unsigned short)(r >> 16);
}

union ABits { u32x4 q[2]; v16bf v; };

// A fragment: 16x32 bf16, row m = lane&15; lane-half h holds k = 16h..16h+15.
__device__ __forceinline__ v16bf load_afrag(const unsigned short* __restrict__ A,
                                            int row, int kc, int lane) {
  const unsigned short* p = A + (size_t)row * RNN_K + kc + ((lane & 16) ? 16 : 0);
  ABits f;
  f.q[0] = *(const u32x4*)(p);
  f.q[1] = *(const u32x4*)(p + 8);
  return f.v;
}

// B fragment from the LDS-staged tile (tile layout: [n(64)][k(32)] bf16).
__device__ __forceinline__ v16bf lds_bfrag(const unsigned short* __restrict__ buf,
                                           int nLocal, int lane) {
  const unsigned short* p = buf + nLocal * 32 + ((lane & 16) ? 16 : 0);
  return *(const v16bf*)p;
}

// CDNA5 async global->LDS copy, 16B per lane; tracked by ASYNCcnt.
__device__ __forceinline__ void async_ld_b128(unsigned lds_off,
                                              const unsigned short* g) {
  unsigned long long ga = (unsigned long long)(size_t)g;
  asm volatile("global_load_async_to_lds_b128 %0, %1, off"
               :: "v"(lds_off), "v"(ga)
               : "memory");
}

#define WMMA_BF16(ACC, AV, BV) \
  (ACC) = __builtin_amdgcn_wmma_f32_16x16x32_bf16(false, (AV), false, (BV), \
                                                  (short)0, (ACC), false, false)

// Compute one staged K-chunk: 4 LDS B-frags up-front, then 4 independent WMMAs.
__device__ __forceinline__ void chunk_mma(v8f acc[4], v16bf a,
                                          const unsigned short* __restrict__ buf,
                                          int nnn, int lane) {
  v16bf b0 = lds_bfrag(buf, 0 * 16 + nnn, lane);
  v16bf b1 = lds_bfrag(buf, 1 * 16 + nnn, lane);
  v16bf b2 = lds_bfrag(buf, 2 * 16 + nnn, lane);
  v16bf b3 = lds_bfrag(buf, 3 * 16 + nnn, lane);
  WMMA_BF16(acc[0], a, b0);
  WMMA_BF16(acc[1], a, b1);
  WMMA_BF16(acc[2], a, b2);
  WMMA_BF16(acc[3], a, b3);
}

// ---------------------------------------------------------------------------
// Block-cooperative GEMM core: 256-thread block, tile M=128 x N=64.
// Wave w accumulates rows [wave*16,+16) x cols [n0,+64) into acc[4].
// B tile staged to LDS with async loads, triple-buffered, ONE barrier/chunk:
//   wait(chunk c) ; barrier ; stage(chunk c+2) ; compute(chunk c)
// The buffer written for c+2 was last read at c-1, ordered by this barrier.
// Last two chunks peeled so the steady-state loop body is branch-free.
// ---------------------------------------------------------------------------
__device__ __forceinline__ void gemm_block(
    v8f acc[4],
    const unsigned short* __restrict__ A, int rowA, bool zeroA,
    const unsigned short* __restrict__ W, int n0,
    unsigned short* Bsh /* [3][BTILE_H] */, int lane, int tid) {
  const int nn = tid >> 2;           // 0..63 : staged col
  const int k8 = (tid & 3) * 8;      // 0,8,16,24 : staged k-offset (8 halves)
  const unsigned short* gsrc = W + (size_t)(n0 + nn) * RNN_K + k8;
  const unsigned ldsBase = (unsigned)(size_t)(Bsh + nn * 32 + k8);
  const int nnn = lane & 15;
  const v16bf zf = {};

  // stage chunks 0 and 1 (ASYNCcnt = 2 outstanding)
  async_ld_b128(ldsBase, gsrc);
  async_ld_b128(ldsBase + BTILE_H * 2, gsrc + 32);

  v16bf a_cur = zf;
  if (!zeroA) a_cur = load_afrag(A, rowA, 0, lane);

  int stageBuf = 2;                  // LDS buffer for chunk c+2
  int compBuf  = 0;                  // LDS buffer for chunk c

  for (int c = 0; c < KCHUNKS - 2; ++c) {
    const int kc = c * 32;
    asm volatile("s_wait_asynccnt 0x1" ::: "memory");   // chunk c landed
    __syncthreads();                                     // all waves: c landed,
                                                         // c-1 reads retired
    async_ld_b128(ldsBase + (unsigned)stageBuf * (BTILE_H * 2), gsrc + kc + 64);

    v16bf a_nxt = zf;
    if (!zeroA) {
      __builtin_prefetch(A + (size_t)rowA * RNN_K + kc + 96, 0, 1);
      a_nxt = load_afrag(A, rowA, kc + 32, lane);
    }
    chunk_mma(acc, a_cur, Bsh + compBuf * BTILE_H, nnn, lane);
    a_cur = a_nxt;
    stageBuf = (stageBuf == 2) ? 0 : stageBuf + 1;
    compBuf  = (compBuf == 2) ? 0 : compBuf + 1;
  }

  // chunk KCHUNKS-2 (nothing left to stage)
  asm volatile("s_wait_asynccnt 0x1" ::: "memory");
  __syncthreads();
  {
    v16bf a_nxt = zf;
    if (!zeroA) a_nxt = load_afrag(A, rowA, (KCHUNKS - 1) * 32, lane);
    chunk_mma(acc, a_cur, Bsh + compBuf * BTILE_H, nnn, lane);
    a_cur = a_nxt;
    compBuf = (compBuf == 2) ? 0 : compBuf + 1;
  }

  // chunk KCHUNKS-1
  asm volatile("s_wait_asynccnt 0x0" ::: "memory");
  __syncthreads();
  chunk_mma(acc, a_cur, Bsh + compBuf * BTILE_H, nnn, lane);

  __syncthreads();                   // Bsh free for a subsequent gemm_block
}

// --------------------------- data-prep kernels -----------------------------

// (B,S,D) fp32 -> time-major (S*B, D) bf16
__global__ __launch_bounds__(256) void convert_input(
    const float* __restrict__ in, unsigned short* __restrict__ Xbf) {
  const size_t idx = ((size_t)blockIdx.x * blockDim.x + threadIdx.x) * 4;
  const int d = (int)(idx & (RNN_K - 1));
  const size_t r = idx >> 10;                // r = t*64 + b
  const int t = (int)(r >> 6);
  const int b = (int)(r & 63);
  f32x4 v = *(const f32x4*)(in + ((size_t)b * RNN_S + t) * RNN_K + d);
  u16x4 o;
  o.x = f2bf(v.x); o.y = f2bf(v.y); o.z = f2bf(v.z); o.w = f2bf(v.w);
  *(u16x4*)(Xbf + idx) = o;
}

__global__ __launch_bounds__(256) void cast_bf16(
    const float* __restrict__ src, unsigned short* __restrict__ dst) {
  const size_t idx = ((size_t)blockIdx.x * blockDim.x + threadIdx.x) * 4;
  f32x4 v = *(const f32x4*)(src + idx);
  u16x4 o;
  o.x = f2bf(v.x); o.y = f2bf(v.y); o.z = f2bf(v.z); o.w = f2bf(v.w);
  *(u16x4*)(dst + idx) = o;
}

__global__ __launch_bounds__(256) void add_bias(
    const float* __restrict__ a, const float* __restrict__ b,
    float* __restrict__ o) {
  int i = blockIdx.x * blockDim.x + threadIdx.x;
  o[i] = a[i] + b[i];
}

// --------------------------- GEMM kernels ----------------------------------

// P[16384,1024] = A(bf16) @ W^T + bias (fp32). Block tile 128x64.
__global__ __launch_bounds__(256) void proj_gemm(
    const unsigned short* __restrict__ A,
    const unsigned short* __restrict__ W,
    const float* __restrict__ bias,
    float* __restrict__ P) {
  __shared__ unsigned short Bsh[3][BTILE_H];
  const int tid  = threadIdx.x;
  const int lane = tid & 31;
  const int wave = tid >> 5;
  const int mBase = (blockIdx.x >> 4) * 128 + wave * 16;
  const int n0    = (blockIdx.x & 15) * 64;
  const int nn    = lane & 15;
  v8f acc[4] = {};

  gemm_block(acc, A, mBase + nn, false, W, n0, &Bsh[0][0], lane, tid);

  const int hh = (lane >> 4) & 1;
#pragma unroll
  for (int s = 0; s < 4; ++s) {
    const int n = n0 + s * 16 + nn;
    const float bv = bias[n];
#pragma unroll
    for (int i = 0; i < 8; ++i) {
      const int m = mBase + i + hh * 8;
      P[(size_t)m * RNN_K + n] = acc[s][i] + bv;
    }
  }
}

// One recurrence step: h(t) = tanh(P[t] + h(t-1) @ Whh^T). 64 waves total,
// latency-bound serial chain -> simple direct-from-global form.
__global__ __launch_bounds__(256) void rnn_step(
    const float* __restrict__ P,
    unsigned short* __restrict__ Hseq,            // read t-1, write t
    const unsigned short* __restrict__ Whh,
    int t,
    float* __restrict__ outF,                     // nullable: d_out fwd rows
    float* __restrict__ hidden, int slotA, int slotB) {
  const int lane  = threadIdx.x & 31;
  const int wave  = (blockIdx.x * blockDim.x + threadIdx.x) >> 5;
  const int mBase = (wave >> 4) * 16;             // rows = batch
  const int nBase = (wave & 15) * 64;
  const int nn    = lane & 15;
  v8f acc[4] = {};

  if (t > 0) {
    const unsigned short* Hprev = Hseq + (size_t)(t - 1) * RNN_B * RNN_K;
    const int row = mBase + nn;
    const int bofs = (lane & 16) ? 16 : 0;
    for (int kc = 0; kc < RNN_K; kc += 32) {
      v16bf a  = load_afrag(Hprev, row, kc, lane);
      v16bf b0 = *(const v16bf*)(Whh + (size_t)(nBase +  0 + nn) * RNN_K + kc + bofs);
      v16bf b1 = *(const v16bf*)(Whh + (size_t)(nBase + 16 + nn) * RNN_K + kc + bofs);
      v16bf b2 = *(const v16bf*)(Whh + (size_t)(nBase + 32 + nn) * RNN_K + kc + bofs);
      v16bf b3 = *(const v16bf*)(Whh + (size_t)(nBase + 48 + nn) * RNN_K + kc + bofs);
      WMMA_BF16(acc[0], a, b0);
      WMMA_BF16(acc[1], a, b1);
      WMMA_BF16(acc[2], a, b2);
      WMMA_BF16(acc[3], a, b3);
    }
  }

  unsigned short* Hcur = Hseq + (size_t)t * RNN_B * RNN_K;
  const int hh = (lane >> 4) & 1;
#pragma unroll
  for (int s = 0; s < 4; ++s) {
    const int n = nBase + s * 16 + nn;
#pragma unroll
    for (int i = 0; i < 8; ++i) {
      const int bb = mBase + i + hh * 8;          // batch row in [0,64)
      float h = tanhf(P[(size_t)(t * RNN_B + bb) * RNN_K + n] + acc[s][i]);
      Hcur[(size_t)bb * RNN_K + n] = f2bf(h);
      if (outF)   outF[(size_t)n * RNN_M + t * RNN_B + bb] = h;  // out[c, t*B+b]
      if (hidden) {
        hidden[(size_t)bb * 4096 + slotA * 1024 + n] = h;
        hidden[(size_t)bb * 4096 + slotB * 1024 + n] = h;        // source bug: hf dup
      }
    }
  }
}

// Backward direction (non-recurrent thanks to the reference bug):
// hb(t) = tanh(xb(t)@Wih^T + hf(t-1)@Whh^T + bias), all t in parallel.
__global__ __launch_bounds__(256) void bwd_fused(
    const unsigned short* __restrict__ A1, int remapXb,
    const unsigned short* __restrict__ Wih,
    const unsigned short* __restrict__ Hf,        // forward sequence (shifted)
    const unsigned short* __restrict__ Whh,
    const float* __restrict__ bias,
    unsigned short* __restrict__ outBf,           // nullable: bf16 sequence
    float* __restrict__ outFlip) {                // nullable: d_out bwd rows (flipped t)
  __shared__ unsigned short Bsh[3][BTILE_H];
  const int tid  = threadIdx.x;
  const int lane = tid & 31;
  const int wave = tid >> 5;
  const int mBase = (blockIdx.x >> 4) * 128 + wave * 16;
  const int n0    = (blockIdx.x & 15) * 64;
  const int nn    = lane & 15;
  const int t     = mBase >> 6;                   // uniform per 16-row wave tile
  const int bBase = mBase & 63;
  v8f acc[4] = {};

  // pass 1: xb(t) @ Wih^T.  xb(t) = input[ t==0 ? 0 : S-t ] (time-major A1).
  int r1;
  if (remapXb) {
    const int sidx = (t == 0) ? 0 : (RNN_S - t);
    r1 = sidx * RNN_B + bBase + nn;
  } else {
    r1 = mBase + nn;
  }
  gemm_block(acc, A1, r1, false, Wih, n0, &Bsh[0][0], lane, tid);

  // pass 2: hf(t-1) @ Whh^T (zero state at t==0; uniform per wave, all waves
  // still execute the block-cooperative staging/barriers).
  const bool z = (t == 0);
  gemm_block(acc, Hf, z ? nn : (mBase - RNN_B + nn), z, Whh, n0,
             &Bsh[0][0], lane, tid);

  const int hh = (lane >> 4) & 1;
#pragma unroll
  for (int s = 0; s < 4; ++s) {
    const int n = n0 + s * 16 + nn;
    const float bv = bias[n];
#pragma unroll
    for (int i = 0; i < 8; ++i) {
      const int bb = bBase + i + hh * 8;
      const float h = tanhf(acc[s][i] + bv);
      const int m = t * RNN_B + bb;
      if (outBf)   outBf[(size_t)m * RNN_K + n] = f2bf(h);
      if (outFlip) outFlip[(size_t)n * RNN_M + (RNN_S - 1 - t) * RNN_B + bb] = h;
    }
  }
}

// --------------------------------- host ------------------------------------

extern "C" void kernel_launch(void* const* d_in, const int* in_sizes, int n_in,
                              void* d_out, int out_size, void* d_ws, size_t ws_size,
                              hipStream_t stream) {
  (void)in_sizes; (void)n_in; (void)out_size; (void)ws_size;  // ws needs ~220 MB

  const float* input = (const float*)d_in[0];
  const float* Wih_f = (const float*)d_in[1];
  const float* Whh_f = (const float*)d_in[2];
  const float* bih_f = (const float*)d_in[3];
  const float* bhh_f = (const float*)d_in[4];
  const float* Wih_b = (const float*)d_in[5];
  const float* Whh_b = (const float*)d_in[6];
  const float* bih_b = (const float*)d_in[7];
  const float* bhh_b = (const float*)d_in[8];

  float* out = (float*)d_out;                     // (2H, S*B) row-major
  float* hid = out + (size_t)2048 * RNN_M;        // (B, 4, H)

  char* ws = (char*)d_ws;
  size_t off = 0;
  auto take = [&](size_t bytes) -> char* {
    char* p = ws + off;
    off += (bytes + 255) & ~(size_t)255;
    return p;
  };
  unsigned short* Xbf = (unsigned short*)take((size_t)RNN_M * RNN_K * 2);
  unsigned short* Wb[8];
  for (int i = 0; i < 8; ++i) Wb[i] = (unsigned short*)take((size_t)1024 * 1024 * 2);
  float* bias[4];
  for (int i = 0; i < 4; ++i) bias[i] = (float*)take((size_t)1024 * 4);
  float*          P   = (float*)take((size_t)RNN_M * RNN_K * 4);
  unsigned short* HF0 = (unsigned short*)take((size_t)RNN_M * RNN_K * 2);
  unsigned short* HF1 = (unsigned short*)take((size_t)RNN_M * RNN_K * 2);
  unsigned short* HB0 = (unsigned short*)take((size_t)RNN_M * RNN_K * 2);

  // --- prep: bf16 input (time-major), bf16 weights (L2-resident), fused biases
  convert_input<<<16384, 256, 0, stream>>>(input, Xbf);
  const float* wsrc[8] = { Wih_f, Wih_f + 1048576, Whh_f, Whh_f + 1048576,
                           Wih_b, Wih_b + 1048576, Whh_b, Whh_b + 1048576 };
  for (int i = 0; i < 8; ++i)
    cast_bf16<<<1024, 256, 0, stream>>>(wsrc[i], Wb[i]);
  add_bias<<<4, 256, 0, stream>>>(bih_f,        bhh_f,        bias[0]);
  add_bias<<<4, 256, 0, stream>>>(bih_f + 1024, bhh_f + 1024, bias[1]);
  add_bias<<<4, 256, 0, stream>>>(bih_b,        bhh_b,        bias[2]);
  add_bias<<<4, 256, 0, stream>>>(bih_b + 1024, bhh_b + 1024, bias[3]);

  // --- forward layer 0: big input projection, then 256 small recurrent steps
  proj_gemm<<<2048, 256, 0, stream>>>(Xbf, Wb[0], bias[0], P);
  for (int t = 0; t < RNN_S; ++t)
    rnn_step<<<8, 256, 0, stream>>>(P, HF0, Wb[2], t, nullptr,
                                    (t == RNN_S - 1) ? hid : nullptr, 0, 2);
  // --- forward layer 1 (writes fwd half of `out` directly)
  proj_gemm<<<2048, 256, 0, stream>>>(HF0, Wb[1], bias[1], P);
  for (int t = 0; t < RNN_S; ++t)
    rnn_step<<<8, 256, 0, stream>>>(P, HF1, Wb[3], t, out,
                                    (t == RNN_S - 1) ? hid : nullptr, 1, 3);
  // --- backward layers: fully parallel fused double-GEMMs
  bwd_fused<<<2048, 256, 0, stream>>>(Xbf, 1, Wb[4], HF0, Wb[6], bias[2],
                                      HB0, nullptr);
  bwd_fused<<<2048, 256, 0, stream>>>(HB0, 0, Wb[5], HF1, Wb[7], bias[3],
                                      nullptr, out + (size_t)1024 * RNN_M);
}
